// SlidingWindowAttention_25838523253122
// MI455X (gfx1250) — matrix operand order repaired and checked by reading, not compiled, not run on hardware
//
#include <hip/hip_runtime.h>
#include <hip/hip_bf16.h>
#include <math.h>
#include <stdint.h>

typedef __attribute__((ext_vector_type(16))) _Float16 v16h;
typedef __attribute__((ext_vector_type(8)))  _Float16 v8h;
typedef __attribute__((ext_vector_type(8)))  float    v8f;
typedef __attribute__((ext_vector_type(4)))  unsigned int u32x4;
typedef __attribute__((ext_vector_type(8)))  unsigned int u32x8;

static constexpr int Bn  = 4;
static constexpr int Sn  = 4096;
static constexpr int Dn  = 1024;
static constexpr int Hn  = 16;
static constexpr int Wn  = 128;
static constexpr int HDn = 64;
static constexpr int nWn = Sn / Wn;        // 32 windows
static constexpr int Mrows = Bn * Sn;      // 16384 token rows

#define WMMA_F16(a, b, c) \
  __builtin_amdgcn_wmma_f32_16x16x32_f16(false, (a), false, (b), (short)0, (c), false, false)

// ---------------------------------------------------------------------------
// WMMA fragment loaders (layouts per CDNA5 ISA 7.12.2, wave32)
// A 16x32 f16: lane = hi*16+m holds row m; elems 0..7 -> K=hi*8+0..7,
//              elems 8..15 -> K=16+hi*8+0..7
// ---------------------------------------------------------------------------
__device__ inline v16h load_a_frag(const _Float16* base, int stride) {
  const int lane = threadIdx.x & 31;
  const int hi = lane >> 4;
  const int m  = lane & 15;
  const _Float16* p = base + m * stride + hi * 8;
  union { v16h v; v8h h[2]; } u;
  u.h[0] = *(const v8h*)(p);
  u.h[1] = *(const v8h*)(p + 16);
  return u.v;
}

// B 32x16 f16 with B[k][n] stored as Row[n][k] ("W-transposed" layout):
// lane = hi*16+n: elems 0..7 -> K=hi*8+0..7, elems 8..15 -> K=16+hi*8+0..7
__device__ inline v16h load_b_frag(const _Float16* base, int stride) {
  const int lane = threadIdx.x & 31;
  const int hi = lane >> 4;
  const int n  = lane & 15;
  const _Float16* p = base + n * stride + hi * 8;
  union { v16h v; v8h h[2]; } u;
  u.h[0] = *(const v8h*)(p);
  u.h[1] = *(const v8h*)(p + 16);
  return u.v;
}

// ---------------------------------------------------------------------------
// Tensor Data Mover: DMA a 2D tile (tile_d0 f16 elems wide, tile_d1 rows,
// row stride stride0 elems) from global memory into LDS at lds_off bytes.
// D# packing per CDNA5 ISA (group0 128b + group1 256b).
// Issued per-wave (EXEC ignored by TDM); tracked with TENSORcnt.
// ---------------------------------------------------------------------------
__device__ inline unsigned lds_offset_of(const void* p) {
  // Generic LDS addresses carry the LDS byte offset in the low 32 bits
  // (ISA 10.2).
  return (unsigned)(uintptr_t)p;
}

__device__ inline void tdm_load_2d(unsigned lds_off, const void* gptr,
                                   unsigned tile_d0, unsigned tile_d1,
                                   unsigned long long stride0_elems) {
  const unsigned long long ga = (unsigned long long)(uintptr_t)gptr;
  const unsigned td0 = 1u << 20;   // tensor dims >> tile dims: no OOB clipping
  const unsigned td1 = 1u << 20;
  u32x4 g0;
  g0[0] = 1u;                                           // count=1, user mode
  g0[1] = lds_off;                                      // lds_addr (bytes)
  g0[2] = (unsigned)(ga & 0xFFFFFFFFu);                 // global_addr[31:0]
  g0[3] = (unsigned)((ga >> 32) & 0x01FFFFFFu)          // global_addr[56:32]
          | 0x80000000u;                                // type=2 ("image")
  u32x8 g1;
  g1[0] = 1u << 16;                                     // data_size=1 (2B)
  g1[1] = (td0 & 0xFFFFu) << 16;                        // tensor_dim0 lo
  g1[2] = ((td0 >> 16) & 0xFFFFu) | ((td1 & 0xFFFFu) << 16);
  g1[3] = ((td1 >> 16) & 0xFFFFu) | (tile_d0 << 16);    // tile_dim0
  g1[4] = tile_d1 & 0xFFFFu;                            // tile_dim1 (dim2=0)
  g1[5] = (unsigned)(stride0_elems & 0xFFFFFFFFu);      // dim0_stride lo
  g1[6] = (unsigned)((stride0_elems >> 32) & 0xFFFFu);  // dim0_stride hi
  g1[7] = 0u;                                           // dim1_stride unused
  asm volatile("tensor_load_to_lds %0, %1" :: "s"(g0), "s"(g1) : "memory");
}

// ---------------------------------------------------------------------------
// fp32 -> f16 conversion (8 elements / thread, n multiple of 2048)
// ---------------------------------------------------------------------------
__global__ __launch_bounds__(256) void cvt_f32_to_f16_kernel(
    const float* __restrict__ src, _Float16* __restrict__ dst) {
  const int i = (blockIdx.x * 256 + threadIdx.x) * 8;
  const float4 a = *(const float4*)(src + i);
  const float4 b = *(const float4*)(src + i + 4);
  v8h o;
  o[0] = (_Float16)a.x; o[1] = (_Float16)a.y;
  o[2] = (_Float16)a.z; o[3] = (_Float16)a.w;
  o[4] = (_Float16)b.x; o[5] = (_Float16)b.y;
  o[6] = (_Float16)b.z; o[7] = (_Float16)b.w;
  *(v8h*)(dst + i) = o;
}

// ---------------------------------------------------------------------------
// GEMM: out[m][n] = sum_k A[m][k] * Wt[n][k] + bias[n]
// A: [M][K] f16, Wt: [N][K] f16.  128x128 block tile; K chunked by 128 with
// TDM double buffering (2 x (32KB A + 32KB B) dynamic LDS = 128KB).
// 8 waves (2x4): each wave owns 4x2 WMMA tiles; 32 WMMAs per K-chunk.
// ---------------------------------------------------------------------------
__global__ __launch_bounds__(256) void gemm_bias_kernel(
    const _Float16* __restrict__ A, const _Float16* __restrict__ Wt,
    const float* __restrict__ bias,
    float* __restrict__ outF32, _Float16* __restrict__ outF16,
    int N, int K) {
  extern __shared__ __align__(32) char gsmem[];
  // Buffer layout (bytes): buf*65536 -> A tile (32KB), +32768 -> B tile (32KB)

  const int m0 = blockIdx.y * 128;
  const int n0 = blockIdx.x * 128;
  const int t  = threadIdx.x;
  const int wave = t >> 5;
  const int wm = wave >> 2;          // 0..1 : 64-row band
  const int wn = wave & 3;           // 0..3 : 32-col band

  const v8f vzero = {0.f, 0.f, 0.f, 0.f, 0.f, 0.f, 0.f, 0.f};
  v8f acc[4][2];
  for (int i = 0; i < 4; ++i)
    for (int j = 0; j < 2; ++j) acc[i][j] = vzero;

  // Prologue: DMA first K-chunk into buffer 0.
  if (t == 0) {
    tdm_load_2d(lds_offset_of(gsmem),         A  + (size_t)m0 * K, 128, 128, K);
    tdm_load_2d(lds_offset_of(gsmem + 32768), Wt + (size_t)n0 * K, 128, 128, K);
    __builtin_amdgcn_s_wait_tensorcnt(0);
  }
  __syncthreads();

  int buf = 0;
  for (int kc = 0; kc < K; kc += 128) {
    char* cur = gsmem + buf * 65536;
    char* nxt = gsmem + (buf ^ 1) * 65536;
    // Issue DMA for the next chunk into the other buffer (overlaps compute).
    if (t == 0 && kc + 128 < K) {
      tdm_load_2d(lds_offset_of(nxt),
                  A  + (size_t)m0 * K + kc + 128, 128, 128, K);
      tdm_load_2d(lds_offset_of(nxt + 32768),
                  Wt + (size_t)n0 * K + kc + 128, 128, 128, K);
    }
    const _Float16* sAb = (const _Float16*)cur;
    const _Float16* sBb = (const _Float16*)(cur + 32768);
    // Consume current buffer: 4 sub-steps of K=32.
    for (int ks = 0; ks < 128; ks += 32) {
      for (int i = 0; i < 4; ++i) {
        const v16h af = load_a_frag(&sAb[(wm * 64 + i * 16) * 128 + ks], 128);
        for (int j = 0; j < 2; ++j) {
          const v16h bf = load_b_frag(&sBb[(wn * 32 + j * 16) * 128 + ks], 128);
          acc[i][j] = WMMA_F16(af, bf, acc[i][j]);
        }
      }
    }
    if (t == 0) __builtin_amdgcn_s_wait_tensorcnt(0);
    __syncthreads();
    buf ^= 1;
  }

  const int lane = t & 31;
  const int hi = lane >> 4;
  const int nn = lane & 15;
  for (int i = 0; i < 4; ++i) {
    for (int j = 0; j < 2; ++j) {
      const int col  = n0 + wn * 32 + j * 16 + nn;
      const float bv = bias[col];
      const int row0 = m0 + wm * 64 + i * 16 + hi * 8;
      for (int r = 0; r < 8; ++r) {
        const float vv = acc[i][j][r] + bv;
        const size_t idx = (size_t)(row0 + r) * N + col;
        if (outF16) outF16[idx] = (_Float16)vv;
        else        outF32[idx] = vv;
      }
    }
  }
}

// ---------------------------------------------------------------------------
// Window attention: one block per (batch, window, head). 256 threads, 8 waves.
// S = (Q K^T) * scale  ->  softmax  ->  O = P V.  All matmuls via WMMA.
// Q/K tiles staged by TDM; V staged transposed by the threads.
// Dynamic LDS: Q(16K) K(16K) Vt(16K) S_f32(64K) P_f16(32K) = 144KB.
// ---------------------------------------------------------------------------
__global__ __launch_bounds__(256) void window_attn_kernel(
    const _Float16* __restrict__ QH, const _Float16* __restrict__ KH,
    const _Float16* __restrict__ VH, _Float16* __restrict__ AO) {
  extern __shared__ __align__(32) char smem[];
  _Float16* sQ  = (_Float16*)smem;                 // [128][64]
  _Float16* sK  = sQ + 128 * 64;                   // [128][64]
  _Float16* sVt = sK + 128 * 64;                   // [64][128] (d-major)
  float*    sS  = (float*)(sVt + 64 * 128);        // [128][128] f32
  _Float16* sP  = (_Float16*)(sS + 128 * 128);     // [128][128] f16

  const int h  = blockIdx.x & 15;
  const int w  = (blockIdx.x >> 4) & 31;
  const int bb = blockIdx.x >> 9;
  const size_t rowBase = ((size_t)bb * Sn + (size_t)w * Wn) * Dn + h * HDn;

  const int t  = threadIdx.x;

  // TDM: Q and K tiles (128 rows x 64 halfs, row stride Dn elems).
  if (t == 0) {
    tdm_load_2d(lds_offset_of(sQ), QH + rowBase, 64, 128, Dn);
    tdm_load_2d(lds_offset_of(sK), KH + rowBase, 64, 128, Dn);
  }

  // V staged transposed by the threads (TDM cannot transpose).
  {
    const int r  = t >> 1;           // token row 0..127
    const int sg = t & 1;            // 32-half segment
    const _Float16* gv = VH + rowBase + (size_t)r * Dn + sg * 32;
    const v16h v0 = *(const v16h*)gv;
    const v16h v1 = *(const v16h*)(gv + 16);
    for (int e = 0; e < 16; ++e) {
      sVt[(sg * 32 + e) * 128 + r]      = v0[e];
      sVt[(sg * 32 + 16 + e) * 128 + r] = v1[e];
    }
  }
  if (t == 0) __builtin_amdgcn_s_wait_tensorcnt(0);
  __syncthreads();

  const int wave = t >> 5;
  const int lane = t & 31;
  const int hi = lane >> 4;
  const int nn = lane & 15;
  const v8f vzero = {0.f, 0.f, 0.f, 0.f, 0.f, 0.f, 0.f, 0.f};

  // ---- Scores: S[i][j] = sum_d Q[i][d] K[j][d], 128x128 over 8 waves ----
  {
    const int wm = wave >> 2;        // 0..1
    const int wn = wave & 3;         // 0..3
    v8f acc[4][2];
    for (int i = 0; i < 4; ++i)
      for (int j = 0; j < 2; ++j) acc[i][j] = vzero;
    for (int k0 = 0; k0 < 64; k0 += 32) {
      for (int i = 0; i < 4; ++i) {
        const v16h af = load_a_frag(&sQ[(wm * 64 + i * 16) * 64 + k0], 64);
        for (int j = 0; j < 2; ++j) {
          const v16h bf = load_b_frag(&sK[(wn * 32 + j * 16) * 64 + k0], 64);
          acc[i][j] = WMMA_F16(af, bf, acc[i][j]);
        }
      }
    }
    const float scale = 0.125f;      // 1/sqrt(64)
    for (int i = 0; i < 4; ++i)
      for (int j = 0; j < 2; ++j)
        for (int rr = 0; rr < 8; ++rr)
          sS[(wm * 64 + i * 16 + hi * 8 + rr) * 128 + wn * 32 + j * 16 + nn] =
              acc[i][j][rr] * scale;
  }
  __syncthreads();

  // ---- Row softmax (threads 0..127, one row each) ----
  if (t < 128) {
    float mx = -3.0e38f;
    for (int j = 0; j < 128; ++j) mx = fmaxf(mx, sS[t * 128 + j]);
    float sum = 0.f;
    for (int j = 0; j < 128; ++j) {
      const float e = __expf(sS[t * 128 + j] - mx);
      sS[t * 128 + j] = e;
      sum += e;
    }
    const float inv = 1.f / sum;
    for (int j = 0; j < 128; ++j)
      sP[t * 128 + j] = (_Float16)(sS[t * 128 + j] * inv);
  }
  __syncthreads();

  // ---- O = P (128x128) x V (128x64); wave handles 16 rows, all 64 cols ----
  {
    v8f acc2[4];
    for (int j = 0; j < 4; ++j) acc2[j] = vzero;
    for (int k0 = 0; k0 < 128; k0 += 32) {
      const v16h af = load_a_frag(&sP[(wave * 16) * 128 + k0], 128);
      for (int j = 0; j < 4; ++j) {
        const v16h bf = load_b_frag(&sVt[(j * 16) * 128 + k0], 128);
        acc2[j] = WMMA_F16(af, bf, acc2[j]);
      }
    }
    for (int j = 0; j < 4; ++j)
      for (int rr = 0; rr < 8; ++rr)
        AO[rowBase + (size_t)(wave * 16 + hi * 8 + rr) * Dn + j * 16 + nn] =
            (_Float16)acc2[j][rr];
  }
}

// ---------------------------------------------------------------------------
// Global attention: one wave32 per (batch, head); 32 global tokens.
// Projections are per-token linear, so gathered rows of QH/KH/VH ARE the
// projected global q/k/v.  Overwrites AO rows at the global indices.
// ---------------------------------------------------------------------------
__global__ __launch_bounds__(32) void global_attn_kernel(
    const _Float16* __restrict__ QH, const _Float16* __restrict__ KH,
    const _Float16* __restrict__ VH, const int* __restrict__ gidx,
    _Float16* __restrict__ AO) {
  __shared__ __align__(32) _Float16 sQ[32 * 64];
  __shared__ __align__(32) _Float16 sK[32 * 64];
  __shared__ __align__(32) _Float16 sVt[64 * 32];
  __shared__ __align__(32) float    sS[32 * 32];
  __shared__ __align__(32) _Float16 sP[32 * 32];
  __shared__ int sIdx[32];

  const int h  = blockIdx.x & 15;
  const int bb = blockIdx.x >> 4;
  const int l  = threadIdx.x;
  const int tok = gidx[l];
  sIdx[l] = tok;
  const size_t base = ((size_t)bb * Sn + tok) * Dn + h * HDn;

  for (int c = 0; c < 64; c += 16) {
    *(v16h*)&sQ[l * 64 + c] = *(const v16h*)&QH[base + c];
    *(v16h*)&sK[l * 64 + c] = *(const v16h*)&KH[base + c];
    const v16h vv = *(const v16h*)&VH[base + c];
    for (int e = 0; e < 16; ++e) sVt[(c + e) * 32 + l] = vv[e];
  }
  __syncthreads();

  const int hi = l >> 4;
  const int nn = l & 15;
  const v8f vzero = {0.f, 0.f, 0.f, 0.f, 0.f, 0.f, 0.f, 0.f};

  // Scores 32x32 (K'=64)
  v8f acc[2][2];
  for (int i = 0; i < 2; ++i)
    for (int j = 0; j < 2; ++j) acc[i][j] = vzero;
  for (int k0 = 0; k0 < 64; k0 += 32) {
    for (int i = 0; i < 2; ++i) {
      const v16h af = load_a_frag(&sQ[(i * 16) * 64 + k0], 64);
      for (int j = 0; j < 2; ++j) {
        const v16h bf = load_b_frag(&sK[(j * 16) * 64 + k0], 64);
        acc[i][j] = WMMA_F16(af, bf, acc[i][j]);
      }
    }
  }
  for (int i = 0; i < 2; ++i)
    for (int j = 0; j < 2; ++j)
      for (int rr = 0; rr < 8; ++rr)
        sS[(i * 16 + hi * 8 + rr) * 32 + j * 16 + nn] = acc[i][j][rr] * 0.125f;
  __syncthreads();

  // Softmax: one row per lane
  {
    float mx = -3.0e38f;
    for (int j = 0; j < 32; ++j) mx = fmaxf(mx, sS[l * 32 + j]);
    float sum = 0.f;
    for (int j = 0; j < 32; ++j) {
      const float e = __expf(sS[l * 32 + j] - mx);
      sS[l * 32 + j] = e;
      sum += e;
    }
    const float inv = 1.f / sum;
    for (int j = 0; j < 32; ++j)
      sP[l * 32 + j] = (_Float16)(sS[l * 32 + j] * inv);
  }
  __syncthreads();

  // O = P (32x32) x V (32x64), K''=32
  v8f acc2[2][4];
  for (int i = 0; i < 2; ++i)
    for (int j = 0; j < 4; ++j) acc2[i][j] = vzero;
  for (int i = 0; i < 2; ++i) {
    const v16h af = load_a_frag(&sP[(i * 16) * 32], 32);
    for (int j = 0; j < 4; ++j) {
      const v16h bf = load_b_frag(&sVt[(j * 16) * 32], 32);
      acc2[i][j] = WMMA_F16(af, bf, acc2[i][j]);
    }
  }
  for (int i = 0; i < 2; ++i)
    for (int j = 0; j < 4; ++j)
      for (int rr = 0; rr < 8; ++rr) {
        const int row = i * 16 + hi * 8 + rr;
        AO[((size_t)bb * Sn + sIdx[row]) * Dn + h * HDn + j * 16 + nn] =
            (_Float16)acc2[i][j][rr];
      }
}

// ---------------------------------------------------------------------------
// Host orchestration
// ---------------------------------------------------------------------------
extern "C" void kernel_launch(void* const* d_in, const int* in_sizes, int n_in,
                              void* d_out, int out_size, void* d_ws, size_t ws_size,
                              hipStream_t stream) {
  (void)in_sizes; (void)n_in; (void)out_size; (void)ws_size;

  const float* q   = (const float*)d_in[0];
  const float* k   = (const float*)d_in[1];
  const float* v   = (const float*)d_in[2];
  const float* Wq  = (const float*)d_in[3];
  const float* bq  = (const float*)d_in[4];
  const float* Wk  = (const float*)d_in[5];
  const float* bk  = (const float*)d_in[6];
  const float* Wv  = (const float*)d_in[7];
  const float* bv  = (const float*)d_in[8];
  const float* Wo  = (const float*)d_in[9];
  const float* bo  = (const float*)d_in[10];
  const int*   gi  = (const int*)d_in[11];

  const size_t nX  = (size_t)Mrows * Dn;   // 16,777,216 elems
  const size_t nWe = (size_t)Dn * Dn;      //  1,048,576 elems

  char* ws = (char*)d_ws;
  size_t off = 0;
  auto carve = [&](size_t bytes) { char* p = ws + off; off += bytes; return p; };
  _Float16* Xq16 = (_Float16*)carve(nX * 2);
  _Float16* Xk16 = (_Float16*)carve(nX * 2);
  _Float16* Xv16 = (_Float16*)carve(nX * 2);
  _Float16* QH   = (_Float16*)carve(nX * 2);
  _Float16* KH   = (_Float16*)carve(nX * 2);
  _Float16* VH   = (_Float16*)carve(nX * 2);
  _Float16* AO   = (_Float16*)carve(nX * 2);
  _Float16* Wq16 = (_Float16*)carve(nWe * 2);
  _Float16* Wk16 = (_Float16*)carve(nWe * 2);
  _Float16* Wv16 = (_Float16*)carve(nWe * 2);
  _Float16* Wo16 = (_Float16*)carve(nWe * 2);

  // 1) fp32 -> f16 conversions
  const int cX = (int)(nX / 2048);   // 8192 blocks
  const int cW = (int)(nWe / 2048);  //  512 blocks
  cvt_f32_to_f16_kernel<<<cX, 256, 0, stream>>>(q, Xq16);
  cvt_f32_to_f16_kernel<<<cX, 256, 0, stream>>>(k, Xk16);
  cvt_f32_to_f16_kernel<<<cX, 256, 0, stream>>>(v, Xv16);
  cvt_f32_to_f16_kernel<<<cW, 256, 0, stream>>>(Wq, Wq16);
  cvt_f32_to_f16_kernel<<<cW, 256, 0, stream>>>(Wk, Wk16);
  cvt_f32_to_f16_kernel<<<cW, 256, 0, stream>>>(Wv, Wv16);
  cvt_f32_to_f16_kernel<<<cW, 256, 0, stream>>>(Wo, Wo16);

  // 2) Q/K/V projections (f16 out), TDM-staged double-buffered GEMM
  dim3 ggrid(Dn / 128, Mrows / 128);     // (8, 128)
  const size_t gemmLds = 131072;         // 2 x (32KB A + 32KB B)
  gemm_bias_kernel<<<ggrid, 256, gemmLds, stream>>>(Xq16, Wq16, bq, nullptr, QH, Dn, Dn);
  gemm_bias_kernel<<<ggrid, 256, gemmLds, stream>>>(Xk16, Wk16, bk, nullptr, KH, Dn, Dn);
  gemm_bias_kernel<<<ggrid, 256, gemmLds, stream>>>(Xv16, Wv16, bv, nullptr, VH, Dn, Dn);

  // 3) Windowed attention -> AO
  const size_t attnLds =
      (size_t)(128 * 64 * 2) * 3 + 128 * 128 * 4 + 128 * 128 * 2;  // 147456 B
  window_attn_kernel<<<dim3(Bn * nWn * Hn), 256, attnLds, stream>>>(QH, KH, VH, AO);

  // 4) Global attention overwrites AO rows at global indices
  global_attn_kernel<<<dim3(Bn * Hn), 32, 0, stream>>>(QH, KH, VH, gi, AO);

  // 5) Output projection (f32 out -> d_out)
  gemm_bias_kernel<<<ggrid, 256, gemmLds, stream>>>(AO, Wo16, bo, (float*)d_out, nullptr, Dn, Dn);
}
